// _ExactIntegerConv2d_42210938585109
// MI455X (gfx1250) — 8-bit matrix operand order repaired and checked by reading, compiled, not run on hardware
//
#include <hip/hip_runtime.h>
#include <hip/hip_bf16.h>
#include <stdint.h>

typedef __attribute__((ext_vector_type(8))) int v8i;
typedef __attribute__((ext_vector_type(4))) int v4i;

#define HWD   56
#define HWHW  (56*56)
#define CIN_  128
#define COUT_ 256
#define BATCH 16
#define NPIX  (BATCH*HWHW)
#define KC    18

#define Q_BYTES   ((size_t)NPIX * CIN_)
#define RS_BYTES  ((size_t)NPIX * 4)
#define C2_BYTES  ((size_t)NPIX * 4)
#define W8_BYTES  ((size_t)16 * KC * 1024)

__global__ __launch_bounds__(256)
void quantize_kernel(const float* __restrict__ x,
                     const float* __restrict__ act_s_p,
                     const float* __restrict__ act_q_p,
                     const float* __restrict__ azp_p,
                     const float* __restrict__ guard_a_p,
                     uint8_t* __restrict__ q,
                     int* __restrict__ rowsum)
{
    int p = blockIdx.x * blockDim.x + threadIdx.x;
    if (p >= NPIX) return;
    const float act_s = act_s_p[0];
    const float act_q = act_q_p[0];
    const float azp   = azp_p[0];
    const float ga    = guard_a_p[0];
    const float zp    = azp * act_s / ga;
    const float hi    = zp + act_q - act_s;
    const float inv_s = 1.0f / act_s;

    const int b   = p / HWHW;
    const int rem = p - b * HWHW;
    const float* xp = x + (size_t)b * CIN_ * HWHW + rem;
    uint8_t* qp = q + ((size_t)p << 7);

    int sum = 0;
    #pragma unroll 4
    for (int c0 = 0; c0 < CIN_; c0 += 4) {
        uint32_t packed = 0;
        #pragma unroll
        for (int e = 0; e < 4; ++e) {
            float v = xp[(size_t)(c0 + e) * HWHW];
            v = fminf(fmaxf(v, zp), hi);
            int iq = (int)rintf((v - zp) * inv_s);
            sum += iq;
            packed |= ((uint32_t)(iq & 0xFF)) << (8 * e);
        }
        *(uint32_t*)(qp + c0) = packed;
    }
    rowsum[p] = sum;
}

__global__ __launch_bounds__(256)
void boxsum_kernel(const int* __restrict__ rowsum, float* __restrict__ c2f)
{
    int p = blockIdx.x * blockDim.x + threadIdx.x;
    if (p >= NPIX) return;
    const int b   = p / HWHW;
    const int rem = p - b * HWHW;
    const int oh  = rem / HWD;
    const int ow  = rem - oh * HWD;
    int s = 0;
    #pragma unroll
    for (int kh = 0; kh < 3; ++kh) {
        int ih = oh + kh - 1;
        if ((unsigned)ih >= HWD) continue;
        #pragma unroll
        for (int kw = 0; kw < 3; ++kw) {
            int iw = ow + kw - 1;
            if ((unsigned)iw >= HWD) continue;
            s += rowsum[b * HWHW + ih * HWD + iw];
        }
    }
    c2f[p] = (float)s;
}

__global__ __launch_bounds__(256)
void wpack_kernel(const float* __restrict__ w, uint8_t* __restrict__ w8)
{
    int t = blockIdx.x * blockDim.x + threadIdx.x;
    if (t >= 16 * KC * 32) return;
    const int lane = t & 31;
    const int kc   = (t >> 5) % KC;
    const int tile = t / (KC * 32);
    const int n    = lane & 15;
    const int half = lane >> 4;
    const int cc   = kc & 1;
    const int tap  = kc >> 1;
    const int kh   = tap / 3, kw = tap % 3;
    const int cout = tile * 16 + n;

    uint32_t frag[8];
    #pragma unroll
    for (int d = 0; d < 8; ++d) {
        const int kbase = (d >> 1) * 16 + half * 8 + (d & 1) * 4;
        uint32_t pk = 0;
        #pragma unroll
        for (int e = 0; e < 4; ++e) {
            const int k   = kbase + e;
            const int cin = cc * 64 + k;
            const int iw8 = (int)rintf(w[(((size_t)cout * CIN_ + cin) * 3 + kh) * 3 + kw]);
            pk |= ((uint32_t)(iw8 & 0xFF)) << (8 * e);
        }
        frag[d] = pk;
    }
    uint32_t* dst = (uint32_t*)(w8 + (((size_t)(tile * KC + kc) * 32) + lane) * 32);
    #pragma unroll
    for (int d = 0; d < 8; ++d) dst[d] = frag[d];
}

__device__ __forceinline__ v8i load_a_frag(const uint8_t* __restrict__ q,
                                           int b, int oh, int ow,
                                           int kh, int kw, int cc, int half)
{
    const int ih = oh + kh - 1;
    const int iw = ow + kw - 1;
    v8i a;
    if (((unsigned)ih < HWD) & ((unsigned)iw < HWD)) {
        const uint8_t* ap = q + (((size_t)(b * HWD + ih) * HWD + iw) << 7)
                              + (cc << 6) + (half << 3);
        uint2 d0 = *(const uint2*)(ap +  0);
        uint2 d1 = *(const uint2*)(ap + 16);
        uint2 d2 = *(const uint2*)(ap + 32);
        uint2 d3 = *(const uint2*)(ap + 48);
        a[0] = (int)d0.x; a[1] = (int)d0.y;
        a[2] = (int)d1.x; a[3] = (int)d1.y;
        a[4] = (int)d2.x; a[5] = (int)d2.y;
        a[6] = (int)d3.x; a[7] = (int)d3.y;
    } else {
        a = v8i{0};
    }
    return a;
}

__global__ __launch_bounds__(128)
void gemm_kernel(const uint8_t* __restrict__ q,
                 const uint8_t* __restrict__ w8,
                 const float* __restrict__ c2f,
                 const float* __restrict__ wzp,
                 const float* __restrict__ fp_scale,
                 const float* __restrict__ bias,
                 const float* __restrict__ guard_a_p,
                 const float* __restrict__ guard_w_p,
                 float* __restrict__ out)
{
    const int lane  = threadIdx.x & 31;
    const int wave  = threadIdx.x >> 5;
    const int Mtile = (blockIdx.x * 4 + wave) * 32;
    const int Nbase = blockIdx.y * 64;

    const int bimg = Mtile / HWHW;
    const int rem0 = Mtile - bimg * HWHW;

    const int row  = lane & 15;
    const int half = lane >> 4;
    const int r0   = rem0 + row;
    const int r1   = rem0 + 16 + row;
    const int oh0  = r0 / HWD, ow0 = r0 - oh0 * HWD;
    const int oh1  = r1 / HWD, ow1 = r1 - oh1 * HWD;

    v8i acc[8] = {v8i{0}, v8i{0}, v8i{0}, v8i{0}, v8i{0}, v8i{0}, v8i{0}, v8i{0}};

    #pragma unroll
    for (int kc = 0; kc < KC; ++kc) {
        const int tap = kc >> 1, cc = kc & 1;
        const int kh  = tap / 3, kw = tap % 3;

        const v8i a0 = load_a_frag(q, bimg, oh0, ow0, kh, kw, cc, half);
        const v8i a1 = load_a_frag(q, bimg, oh1, ow1, kh, kw, cc, half);

        if (kc + 1 < KC) {
            const uint8_t* pf = w8 + ((size_t)((Nbase >> 4) * KC + kc + 1) << 10)
                                   + (lane << 5);
            __builtin_prefetch(pf, 0, 1);
        }

        #pragma unroll
        for (int nt = 0; nt < 4; ++nt) {
            const uint8_t* bp = w8 + ((size_t)(((Nbase >> 4) + nt) * KC + kc) << 10)
                                   + (lane << 5);
            v4i b0 = *(const v4i*)(bp);
            v4i b1 = *(const v4i*)(bp + 16);
            v8i bb;
            bb[0] = b0[0]; bb[1] = b0[1]; bb[2] = b0[2]; bb[3] = b0[3];
            bb[4] = b1[0]; bb[5] = b1[1]; bb[6] = b1[2]; bb[7] = b1[3];
            acc[nt]     = __builtin_amdgcn_wmma_i32_16x16x64_iu8(
                false, a0, true, bb, acc[nt],     false, false);
            acc[4 + nt] = __builtin_amdgcn_wmma_i32_16x16x64_iu8(
                false, a1, true, bb, acc[4 + nt], false, false);
        }
    }

    const float ga   = guard_a_p[0];
    const float gw   = guard_w_p[0];
    const int   ncol = lane & 15;
    const size_t obase = (size_t)bimg * COUT_ * HWHW;

    int   rr[2][8];
    float c2v[2][8];
    #pragma unroll
    for (int r = 0; r < 8; ++r) {
        const int m0 = r + half * 8;
        rr[0][r]  = rem0 + m0;
        rr[1][r]  = rem0 + 16 + m0;
        c2v[0][r] = c2f[bimg * HWHW + rr[0][r]];
        c2v[1][r] = c2f[bimg * HWHW + rr[1][r]];
    }

    #pragma unroll
    for (int nt = 0; nt < 4; ++nt) {
        const int cout = Nbase + nt * 16 + ncol;
        const float wz = wzp[cout];
        const float sc = fp_scale[cout];
        const size_t cb = (size_t)cout * HWHW;
        #pragma unroll
        for (int h2 = 0; h2 < 2; ++h2) {
            #pragma unroll
            for (int r = 0; r < 8; ++r) {
                const int idx = rr[h2][r];
                const float f = ga * (gw * (float)acc[h2 * 4 + nt][r]
                                      + wz * c2v[h2][r]);
                out[obase + cb + idx] = f * sc + bias[cb + idx];
            }
        }
    }
}

extern "C" void kernel_launch(void* const* d_in, const int* in_sizes, int n_in,
                              void* d_out, int out_size, void* d_ws, size_t ws_size,
                              hipStream_t stream)
{
    const float* x        = (const float*)d_in[0];
    const float* weight   = (const float*)d_in[1];
    const float* act_s    = (const float*)d_in[2];
    const float* act_q    = (const float*)d_in[3];
    const float* azp      = (const float*)d_in[4];
    const float* wzp      = (const float*)d_in[5];
    const float* fp_scale = (const float*)d_in[6];
    const float* bias     = (const float*)d_in[7];
    const float* guard_a  = (const float*)d_in[8];
    const float* guard_w  = (const float*)d_in[9];
    float* out = (float*)d_out;

    uint8_t* ws     = (uint8_t*)d_ws;
    uint8_t* q      = ws;
    int*     rowsum = (int*)(ws + Q_BYTES);
    float*   c2f    = (float*)(ws + Q_BYTES + RS_BYTES);
    uint8_t* w8     = ws + Q_BYTES + RS_BYTES + C2_BYTES;

    quantize_kernel<<<(NPIX + 255) / 256, 256, 0, stream>>>(
        x, act_s, act_q, azp, guard_a, q, rowsum);
    boxsum_kernel<<<(NPIX + 255) / 256, 256, 0, stream>>>(rowsum, c2f);
    wpack_kernel<<<(16 * KC * 32 + 255) / 256, 256, 0, stream>>>(weight, w8);
    gemm_kernel<<<dim3(NPIX / 128, COUT_ / 64), 128, 0, stream>>>(
        q, w8, c2f, wzp, fp_scale, bias, guard_a, guard_w, out);
}
